// SelfAttention_57518202028698
// MI455X (gfx1250) — matrix operand order repaired and checked by reading, not compiled, not run on hardware
//
#include <hip/hip_runtime.h>
#include <hip/hip_bf16.h>

// ---------------- types ----------------
typedef __attribute__((ext_vector_type(16))) __bf16 v16bf;
typedef __attribute__((ext_vector_type(8)))  __bf16 v8bf;
typedef __attribute__((ext_vector_type(8)))  float  v8f;

#define N_EMBD 1024
#define N_HEAD 16
#define DKDIM  64
#define SEQ    2048
#define BATCH  2
#define MTOT   (BATCH * SEQ)          // 4096 rows of x flattened

// ---------------- helpers ----------------
__device__ __forceinline__ unsigned short f2bf(float f) {
  unsigned u = __float_as_uint(f);
  u += 0x7fffu + ((u >> 16) & 1u);    // round-to-nearest-even
  return (unsigned short)(u >> 16);
}

__device__ __forceinline__ v8f wmma_bf16f32(v16bf a, v16bf b, v8f c) {
  // (neg_a, A, neg_b, B, c_mod, C, reuse_a, reuse_b)
  return __builtin_amdgcn_wmma_f32_16x16x32_bf16(false, a, false, b, (short)0, c,
                                                 false, false);
}

// A operand (16x32 bf16, M x K): lane l holds row m = l%16.
// lanes 0-15: K = {0..7, 16..23}; lanes 16-31: K = {8..15, 24..31}  (ISA 7.12.2)
__device__ __forceinline__ v16bf load_a_frag(const unsigned short* base, int row,
                                             int ldk, int kk, int lane) {
  int half = (lane >> 4) & 1;
  const unsigned short* p = base + (size_t)(row + (lane & 15)) * ldk + kk + half * 8;
  v8bf lo = *(const v8bf*)(p);
  v8bf hi = *(const v8bf*)(p + 16);
  return __builtin_shufflevector(lo, hi, 0, 1, 2, 3, 4, 5, 6, 7,
                                 8, 9, 10, 11, 12, 13, 14, 15);
}

// B operand (32x16 bf16, K x N): lane l holds col n = l%16.
// lanes 0-15: K = {0..7, 16..23}; lanes 16-31: K = {8..15, 24..31} (same as A).
// "base" is row-major [N, K] (i.e. W[n,k] => B[k,n] = W^T), so per-lane contiguous.
__device__ __forceinline__ v16bf load_b_frag(const unsigned short* base, int row,
                                             int ldk, int kk, int lane) {
  int half = (lane >> 4) & 1;
  const unsigned short* p = base + (size_t)(row + (lane & 15)) * ldk + kk + half * 8;
  v8bf lo = *(const v8bf*)(p);
  v8bf hi = *(const v8bf*)(p + 16);
  return __builtin_shufflevector(lo, hi, 0, 1, 2, 3, 4, 5, 6, 7,
                                 8, 9, 10, 11, 12, 13, 14, 15);
}

// ---------------- kernel 0: fp32 -> bf16 ----------------
__global__ void cvt_f32_bf16(const float* __restrict__ src,
                             unsigned short* __restrict__ dst, int n) {
  int i = blockIdx.x * blockDim.x + threadIdx.x;
  if (i < n) dst[i] = f2bf(src[i]);
}

// ---------------- kernel 1: QKV GEMM + scatter ----------------
// C[m,n] = sum_k X[m,k] * W[n,k];  M=4096, N=3072, K=1024
// Q scaled by dk^-0.5 and stored [b,h,t,dk]; K stored [b,h,t,dk]; V stored
// transposed [b,h,dk,t] so the PV B-operand is lane-contiguous later.
__global__ __launch_bounds__(128) void qkv_gemm(
    const unsigned short* __restrict__ X, const unsigned short* __restrict__ W,
    unsigned short* __restrict__ qws, unsigned short* __restrict__ kws,
    unsigned short* __restrict__ vtws) {
  const int K = N_EMBD;
  int lane = threadIdx.x & 31;
  int wave = threadIdx.x >> 5;            // 4 waves: 2x2
  int m0 = blockIdx.x * 128 + (wave >> 1) * 64;
  int n0 = blockIdx.y * 128 + (wave & 1) * 64;

  v8f zero = {0.f, 0.f, 0.f, 0.f, 0.f, 0.f, 0.f, 0.f};
  v8f acc[4][4];
#pragma unroll
  for (int i = 0; i < 4; ++i)
#pragma unroll
    for (int j = 0; j < 4; ++j) acc[i][j] = zero;

  for (int kk = 0; kk < K; kk += 32) {
    if (kk + 32 < K) {
      __builtin_prefetch(X + (size_t)(m0 + (lane & 15)) * K + kk + 32, 0, 0);
      __builtin_prefetch(W + (size_t)(n0 + (lane & 15)) * K + kk + 32, 0, 0);
    }
    v16bf a[4], b[4];
#pragma unroll
    for (int i = 0; i < 4; ++i) a[i] = load_a_frag(X, m0 + 16 * i, K, kk, lane);
#pragma unroll
    for (int j = 0; j < 4; ++j) b[j] = load_b_frag(W, n0 + 16 * j, K, kk, lane);
#pragma unroll
    for (int i = 0; i < 4; ++i)
#pragma unroll
      for (int j = 0; j < 4; ++j) acc[i][j] = wmma_bf16f32(a[i], b[j], acc[i][j]);
  }

  // epilogue: C layout = VGPR r, lane l -> (m = r + 8*(l>=16), n = l%16)
  int halfsel = (lane >> 4) & 1;
#pragma unroll
  for (int i = 0; i < 4; ++i)
#pragma unroll
    for (int j = 0; j < 4; ++j)
#pragma unroll
      for (int r = 0; r < 8; ++r) {
        int m = m0 + 16 * i + r + 8 * halfsel;
        int n = n0 + 16 * j + (lane & 15);
        float v = acc[i][j][r];
        int b_ = m >> 11;                  // / SEQ
        int t = m & (SEQ - 1);
        int sec = n >> 10;                 // 0=q 1=k 2=v
        int inner = n & (N_EMBD - 1);
        int h = inner >> 6;
        int d = inner & (DKDIM - 1);
        size_t bh = (size_t)(b_ * N_HEAD + h);
        if (sec == 0)
          qws[(bh * SEQ + t) * DKDIM + d] = f2bf(v * 0.125f);   // dk^-0.5
        else if (sec == 1)
          kws[(bh * SEQ + t) * DKDIM + d] = f2bf(v);
        else
          vtws[(bh * DKDIM + d) * SEQ + t] = f2bf(v);
      }
}

// ---------------- kernel 2: flash attention ----------------
// grid: (SEQ/64, BATCH*N_HEAD), block 128 (4 waves x 16 query rows)
__global__ __launch_bounds__(128) void attn_fa(
    const unsigned short* __restrict__ q, const unsigned short* __restrict__ k,
    const unsigned short* __restrict__ vt, unsigned short* __restrict__ ao) {
  __shared__ __align__(16) unsigned short pbuf[4][16][64];   // 8 KB

  int lane = threadIdx.x & 31;
  int wave = threadIdx.x >> 5;
  int halfsel = (lane >> 4) & 1;
  int bh = blockIdx.y;
  int b_ = bh >> 4, h = bh & (N_HEAD - 1);
  int qb = blockIdx.x;
  int q0 = qb * 64 + wave * 16;

  const unsigned short* qp = q + (size_t)bh * SEQ * DKDIM;
  const unsigned short* kp = k + (size_t)bh * SEQ * DKDIM;
  const unsigned short* vp = vt + (size_t)bh * DKDIM * SEQ;

  v16bf qf[2];
#pragma unroll
  for (int ds = 0; ds < 2; ++ds) qf[ds] = load_a_frag(qp, q0, DKDIM, ds * 32, lane);

  v8f zero = {0.f, 0.f, 0.f, 0.f, 0.f, 0.f, 0.f, 0.f};
  v8f o[4];
#pragma unroll
  for (int dt = 0; dt < 4; ++dt) o[dt] = zero;
  float rmax[8], rsum[8];
#pragma unroll
  for (int r = 0; r < 8; ++r) { rmax[r] = -1e30f; rsum[r] = 0.f; }

  for (int kb = 0; kb <= qb; ++kb) {
    int k0 = kb * 64;
    // S = Q K^T  (16 x 64 per wave)
    v8f s[4];
#pragma unroll
    for (int nt = 0; nt < 4; ++nt) {
      v8f a = zero;
#pragma unroll
      for (int ds = 0; ds < 2; ++ds) {
        v16bf kf = load_b_frag(kp, k0 + 16 * nt, DKDIM, ds * 32, lane);
        a = wmma_bf16f32(qf[ds], kf, a);
      }
      s[nt] = a;
    }
    // causal mask on the diagonal block
    if (kb == qb) {
#pragma unroll
      for (int nt = 0; nt < 4; ++nt)
#pragma unroll
        for (int r = 0; r < 8; ++r) {
          int qrow = q0 + r + 8 * halfsel;
          int kcol = k0 + nt * 16 + (lane & 15);
          if (kcol > qrow) s[nt][r] = -1e30f;
        }
    }
    // per-row max (rows live per-VGPR; reduce over the 16-lane half)
    float mloc[8];
#pragma unroll
    for (int r = 0; r < 8; ++r)
      mloc[r] = fmaxf(fmaxf(s[0][r], s[1][r]), fmaxf(s[2][r], s[3][r]));
    for (int off = 1; off < 16; off <<= 1)
#pragma unroll
      for (int r = 0; r < 8; ++r)
        mloc[r] = fmaxf(mloc[r], __shfl_xor(mloc[r], off, 32));
    float alpha[8];
#pragma unroll
    for (int r = 0; r < 8; ++r) {
      float nm = fmaxf(rmax[r], mloc[r]);
      alpha[r] = __expf(rmax[r] - nm);
      rmax[r] = nm;
    }
    // P = exp(S - m), row sums
    float psum[8];
#pragma unroll
    for (int r = 0; r < 8; ++r) psum[r] = 0.f;
#pragma unroll
    for (int nt = 0; nt < 4; ++nt)
#pragma unroll
      for (int r = 0; r < 8; ++r) {
        float p = __expf(s[nt][r] - rmax[r]);
        s[nt][r] = p;
        psum[r] += p;
      }
    for (int off = 1; off < 16; off <<= 1)
#pragma unroll
      for (int r = 0; r < 8; ++r) psum[r] += __shfl_xor(psum[r], off, 32);
#pragma unroll
    for (int r = 0; r < 8; ++r) rsum[r] = rsum[r] * alpha[r] + psum[r];
    // rescale running output
#pragma unroll
    for (int dt = 0; dt < 4; ++dt)
#pragma unroll
      for (int r = 0; r < 8; ++r) o[dt][r] *= alpha[r];
    // bounce P through LDS: C-layout -> A-layout
#pragma unroll
    for (int nt = 0; nt < 4; ++nt)
#pragma unroll
      for (int r = 0; r < 8; ++r)
        pbuf[wave][r + 8 * halfsel][nt * 16 + (lane & 15)] = f2bf(s[nt][r]);
    __syncthreads();
    v16bf pa[2];
#pragma unroll
    for (int ds = 0; ds < 2; ++ds) {
      const unsigned short* pp = &pbuf[wave][lane & 15][ds * 32 + halfsel * 8];
      v8bf lo = *(const v8bf*)(pp);
      v8bf hi = *(const v8bf*)(pp + 16);
      pa[ds] = __builtin_shufflevector(lo, hi, 0, 1, 2, 3, 4, 5, 6, 7,
                                       8, 9, 10, 11, 12, 13, 14, 15);
    }
    // O += P V   (V^T stored [dk, t] so B-frag per-lane load is contiguous)
#pragma unroll
    for (int dt = 0; dt < 4; ++dt)
#pragma unroll
      for (int ds = 0; ds < 2; ++ds) {
        v16bf vf = load_b_frag(vp, dt * 16, SEQ, k0 + ds * 32, lane);
        o[dt] = wmma_bf16f32(pa[ds], vf, o[dt]);
      }
    __syncthreads();
  }

  // normalize, store bf16 attention output in [b, t, c] for the proj GEMM
  float inv[8];
#pragma unroll
  for (int r = 0; r < 8; ++r) inv[r] = 1.0f / rsum[r];
#pragma unroll
  for (int dt = 0; dt < 4; ++dt)
#pragma unroll
    for (int r = 0; r < 8; ++r) {
      int tkn = q0 + r + 8 * halfsel;
      int c = h * DKDIM + dt * 16 + (lane & 15);
      ao[((size_t)b_ * SEQ + tkn) * N_EMBD + c] = f2bf(o[dt][r] * inv[r]);
    }
}

// ---------------- kernel 3: output projection ----------------
// out[m,n] = sum_k AO[m,k] * Wp[n,k];  M=4096, N=K=1024, f32 output
__global__ __launch_bounds__(128) void proj_gemm(
    const unsigned short* __restrict__ X, const unsigned short* __restrict__ W,
    float* __restrict__ O) {
  const int K = N_EMBD;
  int lane = threadIdx.x & 31;
  int wave = threadIdx.x >> 5;
  int m0 = blockIdx.x * 128 + (wave >> 1) * 64;
  int n0 = blockIdx.y * 128 + (wave & 1) * 64;

  v8f zero = {0.f, 0.f, 0.f, 0.f, 0.f, 0.f, 0.f, 0.f};
  v8f acc[4][4];
#pragma unroll
  for (int i = 0; i < 4; ++i)
#pragma unroll
    for (int j = 0; j < 4; ++j) acc[i][j] = zero;

  for (int kk = 0; kk < K; kk += 32) {
    if (kk + 32 < K) {
      __builtin_prefetch(X + (size_t)(m0 + (lane & 15)) * K + kk + 32, 0, 0);
      __builtin_prefetch(W + (size_t)(n0 + (lane & 15)) * K + kk + 32, 0, 0);
    }
    v16bf a[4], b[4];
#pragma unroll
    for (int i = 0; i < 4; ++i) a[i] = load_a_frag(X, m0 + 16 * i, K, kk, lane);
#pragma unroll
    for (int j = 0; j < 4; ++j) b[j] = load_b_frag(W, n0 + 16 * j, K, kk, lane);
#pragma unroll
    for (int i = 0; i < 4; ++i)
#pragma unroll
      for (int j = 0; j < 4; ++j) acc[i][j] = wmma_bf16f32(a[i], b[j], acc[i][j]);
  }

  int halfsel = (lane >> 4) & 1;
#pragma unroll
  for (int i = 0; i < 4; ++i)
#pragma unroll
    for (int j = 0; j < 4; ++j)
#pragma unroll
      for (int r = 0; r < 8; ++r) {
        int m = m0 + 16 * i + r + 8 * halfsel;
        int n = n0 + 16 * j + (lane & 15);
        O[(size_t)m * N_EMBD + n] = acc[i][j][r];
      }
}

// ---------------- host launcher ----------------
extern "C" void kernel_launch(void* const* d_in, const int* in_sizes, int n_in,
                              void* d_out, int out_size, void* d_ws, size_t ws_size,
                              hipStream_t stream) {
  (void)in_sizes; (void)n_in; (void)out_size; (void)ws_size;
  const float* x      = (const float*)d_in[0];
  const float* w_attn = (const float*)d_in[1];
  const float* w_proj = (const float*)d_in[2];
  float* out = (float*)d_out;

  // workspace carve-up (all bf16 stored as ushort); total = 48 MiB
  unsigned short* ws  = (unsigned short*)d_ws;
  const size_t NX  = (size_t)MTOT * N_EMBD;          // 4M
  const size_t NWA = (size_t)3 * N_EMBD * N_EMBD;    // 3M
  const size_t NWP = (size_t)N_EMBD * N_EMBD;        // 1M
  unsigned short* xb   = ws;
  unsigned short* wab  = xb  + NX;
  unsigned short* wpb  = wab + NWA;
  unsigned short* qws  = wpb + NWP;
  unsigned short* kws  = qws + NX;
  unsigned short* vtws = kws + NX;
  unsigned short* ao   = vtws + NX;

  // stage 0: down-convert fp32 -> bf16
  cvt_f32_bf16<<<(int)((NX  + 255) / 256), 256, 0, stream>>>(x, xb, (int)NX);
  cvt_f32_bf16<<<(int)((NWA + 255) / 256), 256, 0, stream>>>(w_attn, wab, (int)NWA);
  cvt_f32_bf16<<<(int)((NWP + 255) / 256), 256, 0, stream>>>(w_proj, wpb, (int)NWP);

  // stage 1: qkv = x @ w_attn^T, scattered to [b,h,t,dk] (+ V transposed)
  dim3 g1(MTOT / 128, (3 * N_EMBD) / 128);
  qkv_gemm<<<g1, 128, 0, stream>>>(xb, wab, qws, kws, vtws);

  // stage 2: causal flash attention
  dim3 g2(SEQ / 64, BATCH * N_HEAD);
  attn_fa<<<g2, 128, 0, stream>>>(qws, kws, vtws, ao);

  // stage 3: out = ao @ w_proj^T (f32)
  dim3 g3(MTOT / 128, N_EMBD / 128);
  proj_gemm<<<g3, 128, 0, stream>>>(ao, wpb, out);
}
